// MALBlock_51110110822851
// MI455X (gfx1250) — matrix operand order repaired and checked by reading, not compiled, not run on hardware
//
#include <hip/hip_runtime.h>
#include <hip/hip_bf16.h>

typedef __attribute__((ext_vector_type(16))) __bf16 v16bf;
typedef __attribute__((ext_vector_type(8)))  __bf16 v8bf;
typedef __attribute__((ext_vector_type(8)))  float  v8f;

#define GF_OUT_BF16 1
#define GF_GELU     2
#define GF_BIAS     4

__device__ __forceinline__ unsigned short f2bf(float f) {
  unsigned int u = __float_as_uint(f);
  u += 0x7FFFu + ((u >> 16) & 1u);   // round-to-nearest-even
  return (unsigned short)(u >> 16);
}
__device__ __forceinline__ float bf2f(unsigned short h) {
  return __uint_as_float(((unsigned int)h) << 16);
}
__device__ __forceinline__ v8f wmma_bf16(v16bf a, v16bf b, v8f c) {
  return __builtin_amdgcn_wmma_f32_16x16x32_bf16(false, a, false, b, (short)0, c, false, false);
}
// branch-free tanh-approx GELU
__device__ __forceinline__ float gelu_f(float x) {
  float u = 0.7978845608028654f * (x + 0.044715f * x * x * x);
  float e = __expf(2.f * u);
  float t = 1.f - 2.f / (e + 1.f);
  return 0.5f * x * (1.f + t);
}

// ---------------------------------------------------------------------------
// elementwise helpers
// ---------------------------------------------------------------------------
__global__ void k_f32_to_bf16(const float* __restrict__ in,
                              unsigned short* __restrict__ out, int n) {
  int i = blockIdx.x * blockDim.x + threadIdx.x;
  if (i < n) out[i] = f2bf(in[i]);
}

// in [Bn,R,C] f32 -> out [Bn,C,R] bf16  (pre-transpose for GEMM B operands)
__global__ void k_cvtT_f32_bf16(const float* __restrict__ in,
                                unsigned short* __restrict__ out,
                                int Bn, int R, int C) {
  long i = (long)blockIdx.x * blockDim.x + threadIdx.x;
  long total = (long)Bn * R * C;
  if (i < total) {
    int c = (int)(i % C);
    long t = i / C;
    int r = (int)(t % R);
    int b = (int)(t / R);
    out[((long)b * C + c) * R + r] = f2bf(in[i]);
  }
}

// out [Bn,C,R] bf16 = transpose(a - b), a,b [Bn,R,C] f32
__global__ void k_subT_f32_bf16(const float* __restrict__ a, const float* __restrict__ b,
                                unsigned short* __restrict__ out, int Bn, int R, int C) {
  long i = (long)blockIdx.x * blockDim.x + threadIdx.x;
  long total = (long)Bn * R * C;
  if (i < total) {
    int c = (int)(i % C);
    long t = i / C;
    int r = (int)(t % R);
    int bb = (int)(t / R);
    out[((long)bb * C + c) * R + r] = f2bf(a[i] - b[i]);
  }
}

// in [B,R,C] bf16 -> out [B,C,R] bf16
__global__ void k_transpose_bf16(const unsigned short* __restrict__ in,
                                 unsigned short* __restrict__ out,
                                 int Bn, int R, int C) {
  long i = (long)blockIdx.x * blockDim.x + threadIdx.x;
  long total = (long)Bn * R * C;
  if (i < total) {
    int c = (int)(i % C);
    long t = i / C;
    int r = (int)(t % R);
    int b = (int)(t / R);
    out[((long)b * C + c) * R + r] = in[i];
  }
}

__global__ void k_mem_update(const float* __restrict__ grad, const float* __restrict__ W,
                             const float* __restrict__ S, float* __restrict__ Wn,
                             float* __restrict__ Sn, int n) {
  int i = blockIdx.x * blockDim.x + threadIdx.x;
  if (i < n) {
    float s = 0.9f * S[i] - 0.1f * grad[i];
    Sn[i] = s;
    Wn[i] = 0.98f * W[i] + s;
  }
}

__global__ __launch_bounds__(256) void k_layernorm_bf16(
    const float* __restrict__ x, const float* __restrict__ g,
    const float* __restrict__ b, unsigned short* __restrict__ out, int D) {
  __shared__ float red[256];
  const int row = blockIdx.x, tid = threadIdx.x;
  const float* xr = x + (long)row * D;
  float s = 0.f, s2 = 0.f;
  for (int i = tid; i < D; i += 256) { float v = xr[i]; s += v; s2 += v * v; }
  red[tid] = s; __syncthreads();
  for (int o = 128; o > 0; o >>= 1) { if (tid < o) red[tid] += red[tid + o]; __syncthreads(); }
  float mean = red[0] / D; __syncthreads();
  red[tid] = s2; __syncthreads();
  for (int o = 128; o > 0; o >>= 1) { if (tid < o) red[tid] += red[tid + o]; __syncthreads(); }
  float var = red[0] / D - mean * mean;
  float inv = rsqrtf(var + 1e-5f);
  unsigned short* orow = out + (long)row * D;
  for (int i = tid; i < D; i += 256)
    orow[i] = f2bf((xr[i] - mean) * inv * g[i] + b[i]);
}

// ---------------------------------------------------------------------------
// generic batched bf16 GEMM:  C[b] = act(A[b] @ B[b]^T-stored + bias)
// A  [M,K] row-major bf16 ; Bt [N,K] row-major bf16 (pre-transposed B).
// M%128==0, N%128==0, K%32==0. 256 threads = 8 waves, block tile 128x128,
// wave tile 64x32 (4x2 WMMA frags). Double-buffered LDS, one barrier/step.
// ---------------------------------------------------------------------------
template <int FLAGS>
__global__ __launch_bounds__(256) void k_gemm_bf16(
    const unsigned short* __restrict__ A, const unsigned short* __restrict__ Bt,
    const float* __restrict__ bias, void* __restrict__ Cout,
    int M, int N, int K, long sA, long sB, long sC) {
  __shared__ unsigned short As[2][128 * 40];  // [m][k], padded stride 40
  __shared__ unsigned short Bs[2][128 * 40];  // [n][k], padded stride 40
  const int bz = blockIdx.z;
  const int m0 = blockIdx.y * 128, n0 = blockIdx.x * 128;
  const int tid = threadIdx.x;
  const int lane = tid & 31, wid = tid >> 5;
  const int half = lane >> 4, lrow = lane & 15;
  const int wm = wid >> 2, wn = wid & 3;   // 2x4 wave grid
  v8f acc[4][2];
#pragma unroll
  for (int i = 0; i < 4; i++)
#pragma unroll
    for (int j = 0; j < 2; j++) { v8f z = {0.f,0.f,0.f,0.f,0.f,0.f,0.f,0.f}; acc[i][j] = z; }

  // tile loader: 2 threads/row, 16 contiguous k (32B) each, same for A and Bt
  const int trow = tid >> 1, tkc = (tid & 1) * 16;
  const unsigned short* aRow = A + (long)bz * sA + (long)(m0 + trow) * K + tkc;
  const unsigned short* bRow = Bt + (long)bz * sB + (long)(n0 + trow) * K + tkc;
  unsigned short* aDst = &As[0][trow * 40 + tkc];
  unsigned short* bDst = &Bs[0][trow * 40 + tkc];
  const int ldsBuf = 128 * 40;

  auto loadTile = [&](int buf, int k0) {
    const uint4* sa = (const uint4*)(aRow + k0);
    uint4* da = (uint4*)(aDst + buf * ldsBuf);
    da[0] = sa[0]; da[1] = sa[1];
    const uint4* sb = (const uint4*)(bRow + k0);
    uint4* db = (uint4*)(bDst + buf * ldsBuf);
    db[0] = sb[0]; db[1] = sb[1];
    if (k0 + 64 < K) {                       // prime L2 two tiles ahead
      __builtin_prefetch(aRow + k0 + 64, 0, 1);
      __builtin_prefetch(bRow + k0 + 64, 0, 1);
    }
  };

  const int nIter = K >> 5;
  loadTile(0, 0);
  __syncthreads();
  for (int it = 0; it < nIter; ++it) {
    const int cur = it & 1;
    if (it + 1 < nIter) loadTile(cur ^ 1, (it + 1) << 5);
    const unsigned short* Asb = As[cur];
    const unsigned short* Bsb = Bs[cur];
    v16bf af[4];
#pragma unroll
    for (int i = 0; i < 4; i++) {
      // 16-bit A frag: lanes 0-15 K=0..7 & 16..23, lanes 16-31 K=8..15 & 24..31
      const unsigned short* pr = &Asb[(wm * 64 + i * 16 + lrow) * 40];
      v16bf a;
      ((v8bf*)&a)[0] = *(const v8bf*)(pr + half * 8);
      ((v8bf*)&a)[1] = *(const v8bf*)(pr + 16 + half * 8);
      af[i] = a;
    }
#pragma unroll
    for (int j = 0; j < 2; j++) {
      // B frag: lane column n, lanes 0-15 K=0..7 & 16..23, lanes 16-31 K=8..15 & 24..31
      const unsigned short* pc = &Bsb[(wn * 32 + j * 16 + lrow) * 40];
      v16bf bf;
      ((v8bf*)&bf)[0] = *(const v8bf*)(pc + half * 8);
      ((v8bf*)&bf)[1] = *(const v8bf*)(pc + 16 + half * 8);
#pragma unroll
      for (int i = 0; i < 4; i++)
        acc[i][j] = wmma_bf16(af[i], bf, acc[i][j]);
    }
    __syncthreads();
  }
  // epilogue: C frag layout = VGPR v -> row v + 8*half, col = lane&15
#pragma unroll
  for (int i = 0; i < 4; i++)
#pragma unroll
    for (int j = 0; j < 2; j++) {
      int col = n0 + wn * 32 + j * 16 + lrow;
      float bv = 0.f;
      if constexpr ((FLAGS & GF_BIAS) != 0) bv = bias[col];
#pragma unroll
      for (int v = 0; v < 8; v++) {
        int row = m0 + wm * 64 + i * 16 + v + half * 8;
        float val = acc[i][j][v] + bv;
        if constexpr ((FLAGS & GF_GELU) != 0) val = gelu_f(val);
        long idx = (long)bz * sC + (long)row * N + col;
        if constexpr ((FLAGS & GF_OUT_BF16) != 0) ((unsigned short*)Cout)[idx] = f2bf(val);
        else ((float*)Cout)[idx] = val;
      }
    }
}

// ---------------------------------------------------------------------------
// Flash attention: qkv bf16 [B,L,3072] (q|k|v, 16 heads x 64), out bf16 [B,L,1024]
// grid (L/128, H, B), 256 threads = 8 waves, 16 query rows/wave, 32-key steps,
// double-buffered K/V tiles. Row-sums tracked by a 5th PV WMMA tile whose
// B-fragment is a constant ones-column (no LDS, no shuffle reduction).
// ---------------------------------------------------------------------------
__global__ __launch_bounds__(256) void k_flash_attn(
    const unsigned short* __restrict__ qkv, unsigned short* __restrict__ obuf, int L) {
  __shared__ unsigned short Ks[2][32 * 72];     // [key][dh], stride 72
  __shared__ unsigned short Vts[2][64 * 40];    // [dh][key], stride 40
  __shared__ unsigned short Ps[8 * 16 * 40];    // per-wave P 16x32, stride 40
  const int b = blockIdx.z, h = blockIdx.y;
  const long bL = (long)b * L;
  const int tid = threadIdx.x, lane = tid & 31, wid = tid >> 5;
  const int half = lane >> 4, lrow = lane & 15;
  const int q0 = blockIdx.x * 128 + wid * 16;

  // Q fragments (dh chunks 0..31, 32..63); scale 1/sqrt(64) folded in once
  v16bf qf[2];
  {
    const unsigned short* qrow = qkv + (bL + q0 + lrow) * 3072 + h * 64;
#pragma unroll
    for (int c = 0; c < 2; c++) {
      union { v16bf v; unsigned short s[16]; } u;
      ((v8bf*)&u.v)[0] = *(const v8bf*)(qrow + c * 32 + half * 8);
      ((v8bf*)&u.v)[1] = *(const v8bf*)(qrow + c * 32 + 16 + half * 8);
#pragma unroll
      for (int e = 0; e < 16; e++) u.s[e] = f2bf(bf2f(u.s[e]) * 0.125f);
      qf[c] = u.v;
    }
  }
  // constant B-fragment: column 0 = 1.0, other columns 0 (row-sum accumulator)
  v16bf onesb;
  {
    union { v16bf v; unsigned short s[16]; } u;
    unsigned short val = (lrow == 0) ? (unsigned short)0x3F80 : (unsigned short)0;
#pragma unroll
    for (int e = 0; e < 16; e++) u.s[e] = val;
    onesb = u.v;
  }

  float mrow[8];
  v8f O[5];                                    // 4 dh tiles + 1 row-sum tile
#pragma unroll
  for (int v = 0; v < 8; v++) mrow[v] = -1e30f;
#pragma unroll
  for (int nt = 0; nt < 5; nt++) { v8f z = {0,0,0,0,0,0,0,0}; O[nt] = z; }

  const int lkey = tid >> 3, lc8 = (tid & 7) * 8;
  const unsigned short* kBase = qkv + (bL + lkey) * 3072 + 1024 + h * 64 + lc8;
  const unsigned short* vBase = qkv + (bL + lkey) * 3072 + 2048 + h * 64 + lc8;
  unsigned short* Pw = &Ps[wid * 16 * 40];

  auto loadKV = [&](int buf, int kb) {
    *(v8bf*)(&Ks[buf][lkey * 72 + lc8]) = *(const v8bf*)(kBase + (long)kb * 3072);
    v8bf vv = *(const v8bf*)(vBase + (long)kb * 3072);
    const unsigned short* vs = (const unsigned short*)&vv;
#pragma unroll
    for (int i = 0; i < 8; i++) Vts[buf][(lc8 + i) * 40 + lkey] = vs[i];
    if (kb + 64 < L) {
      __builtin_prefetch(kBase + (long)(kb + 64) * 3072, 0, 1);
      __builtin_prefetch(vBase + (long)(kb + 64) * 3072, 0, 1);
    }
  };

  const int nIter = L >> 5;
  loadKV(0, 0);
  __syncthreads();
  for (int it = 0; it < nIter; ++it) {
    const int cur = it & 1;
    if (it + 1 < nIter) loadKV(cur ^ 1, (it + 1) << 5);

    // S = (Q/8) K^T : two 16-key subtiles, K-dim = dh (2 x 32)
    v8f S[2];
    { v8f z = {0,0,0,0,0,0,0,0}; S[0] = z; S[1] = z; }
#pragma unroll
    for (int s = 0; s < 2; s++) {
#pragma unroll
      for (int c = 0; c < 2; c++) {
        const unsigned short* kp = &Ks[cur][(s * 16 + lrow) * 72 + c * 32];
        v16bf bf;
        ((v8bf*)&bf)[0] = *(const v8bf*)(kp + half * 8);
        ((v8bf*)&bf)[1] = *(const v8bf*)(kp + 16 + half * 8);
        S[s] = wmma_bf16(qf[c], bf, S[s]);
      }
    }
    // online softmax: only the row-MAX needs a cross-lane reduction
    float corr[8];
#pragma unroll
    for (int v = 0; v < 8; v++) {
      float r = fmaxf(S[0][v], S[1][v]);
      r = fmaxf(r, __shfl_xor(r, 1, 32));
      r = fmaxf(r, __shfl_xor(r, 2, 32));
      r = fmaxf(r, __shfl_xor(r, 4, 32));
      r = fmaxf(r, __shfl_xor(r, 8, 32));
      float nm = fmaxf(mrow[v], r);
      corr[v] = __expf(mrow[v] - nm);
      mrow[v] = nm;
      float p0 = __expf(S[0][v] - nm), p1 = __expf(S[1][v] - nm);
      int prow = v + half * 8;
      Pw[prow * 40 + lrow] = f2bf(p0);        // keys 0..15 of chunk
      Pw[prow * 40 + 16 + lrow] = f2bf(p1);   // keys 16..31
    }
    // P as A fragment (same-wave LDS, DS ops are wave-ordered: no barrier needed)
    v16bf pf;
    {
      const unsigned short* pr = &Pw[lrow * 40];
      ((v8bf*)&pf)[0] = *(const v8bf*)(pr + half * 8);
      ((v8bf*)&pf)[1] = *(const v8bf*)(pr + 16 + half * 8);
    }
    // O = diag(corr) O + P [V | 1]
#pragma unroll
    for (int nt = 0; nt < 5; nt++) {
      v16bf bf;
      if (nt < 4) {
        const unsigned short* vp = &Vts[cur][(nt * 16 + lrow) * 40];
        ((v8bf*)&bf)[0] = *(const v8bf*)(vp + half * 8);
        ((v8bf*)&bf)[1] = *(const v8bf*)(vp + 16 + half * 8);
      } else {
        bf = onesb;
      }
      v8f o = O[nt];
#pragma unroll
      for (int v = 0; v < 8; v++) o[v] *= corr[v];
      O[nt] = wmma_bf16(pf, bf, o);
    }
    __syncthreads();
  }
  // row sums live in column 0 of tile 4 (lane 0 / lane 16): broadcast per half
  float lsum[8];
#pragma unroll
  for (int v = 0; v < 8; v++) {
    float s = __shfl(O[4][v], lane & 16, 32);
    lsum[v] = 1.f / s;
  }
  // normalize + write o[b, q, h*64+dh] as bf16
#pragma unroll
  for (int nt = 0; nt < 4; nt++)
#pragma unroll
    for (int v = 0; v < 8; v++) {
      int q = q0 + v + half * 8;
      float val = O[nt][v] * lsum[v];
      obuf[(bL + q) * 1024 + h * 64 + nt * 16 + lrow] = f2bf(val);
    }
}

// ---------------------------------------------------------------------------
extern "C" void kernel_launch(void* const* d_in, const int* in_sizes, int n_in,
                              void* d_out, int out_size, void* d_ws, size_t ws_size,
                              hipStream_t stream) {
  (void)in_sizes; (void)n_in; (void)out_size; (void)ws_size;
  const int B = 2, L = 2048, D = 1024, DK = 128, DV = 128, MLP = 4096;
  const int BL = B * L;

  const float* x     = (const float*)d_in[0];
  const float* Wmem  = (const float*)d_in[1];
  const float* Smem  = (const float*)d_in[2];
  const float* Wk    = (const float*)d_in[3];
  const float* bk    = (const float*)d_in[4];
  const float* Wv    = (const float*)d_in[5];
  const float* bv    = (const float*)d_in[6];
  const float* Wq    = (const float*)d_in[7];
  const float* bq    = (const float*)d_in[8];
  const float* Wu    = (const float*)d_in[9];
  const float* bu    = (const float*)d_in[10];
  const float* ln1_g = (const float*)d_in[11];
  const float* ln1_b = (const float*)d_in[12];
  const float* Wqkv  = (const float*)d_in[13];
  const float* Wo    = (const float*)d_in[14];
  const float* bo    = (const float*)d_in[15];
  const float* ln2_g = (const float*)d_in[16];
  const float* ln2_b = (const float*)d_in[17];
  const float* W1    = (const float*)d_in[18];
  const float* b1    = (const float*)d_in[19];
  const float* W2    = (const float*)d_in[20];
  const float* b2    = (const float*)d_in[21];

  char* ws = (char*)d_ws;
  size_t off = 0;
  auto alloc = [&](size_t bytes) -> void* {
    off = (off + 255) & ~(size_t)255;
    void* p = ws + off;
    off += bytes;
    return p;
  };
  typedef unsigned short u16;
  u16* xb     = (u16*)alloc((size_t)BL * D * 2);
  u16* WkT    = (u16*)alloc((size_t)D * DK * 2);
  u16* WvT    = (u16*)alloc((size_t)D * DV * 2);
  u16* WqT    = (u16*)alloc((size_t)D * DK * 2);
  u16* WuT    = (u16*)alloc((size_t)DV * D * 2);
  u16* WqkvT  = (u16*)alloc((size_t)D * 3072 * 2);
  u16* WoT    = (u16*)alloc((size_t)D * D * 2);
  u16* W1T    = (u16*)alloc((size_t)D * MLP * 2);
  u16* W2T    = (u16*)alloc((size_t)MLP * D * 2);
  u16* WmT    = (u16*)alloc((size_t)B * DK * DV * 2);
  u16* Kb     = (u16*)alloc((size_t)BL * DK * 2);
  u16* Qb     = (u16*)alloc((size_t)BL * DK * 2);
  float* Vf   = (float*)alloc((size_t)BL * DV * 4);
  u16* QWb    = (u16*)alloc((size_t)BL * DV * 2);
  float* xmem = (float*)alloc((size_t)BL * D * 4);
  u16* hb     = (u16*)alloc((size_t)BL * D * 2);
  u16* qkvb   = (u16*)alloc((size_t)BL * 3072 * 2);
  u16* ob     = (u16*)alloc((size_t)BL * D * 2);
  float* xatt = (float*)alloc((size_t)BL * D * 4);
  u16* h2b    = (u16*)alloc((size_t)BL * D * 2);
  u16* m1b    = (u16*)alloc((size_t)BL * MLP * 2);
  float* predf= (float*)alloc((size_t)BL * DV * 4);
  u16* EbT    = (u16*)alloc((size_t)BL * DV * 2);   // [B, DV, L]
  u16* KTb    = (u16*)alloc((size_t)BL * DK * 2);   // [B, DK, L]
  float* gradf= (float*)alloc((size_t)B * DK * DV * 4);

  auto cvtT = [&](const float* in, u16* out, int Bn, int R, int C) {
    long n = (long)Bn * R * C;
    k_cvtT_f32_bf16<<<(unsigned)((n + 255) / 256), 256, 0, stream>>>(in, out, Bn, R, C);
  };
  k_f32_to_bf16<<<(BL * D + 255) / 256, 256, 0, stream>>>(x, xb, BL * D);
  cvtT(Wk, WkT, 1, D, DK);      cvtT(Wv, WvT, 1, D, DV);    cvtT(Wq, WqT, 1, D, DK);
  cvtT(Wu, WuT, 1, DV, D);      cvtT(Wqkv, WqkvT, 1, D, 3072);
  cvtT(Wo, WoT, 1, D, D);       cvtT(W1, W1T, 1, D, MLP);   cvtT(W2, W2T, 1, MLP, D);
  cvtT(Wmem, WmT, B, DK, DV);

  auto gemm = [&](const u16* A, const u16* Bt, const float* bias, void* C,
                  int M, int N, int K, long sA, long sB, long sC, int batch, int flags) {
    dim3 g(N / 128, M / 128, batch);
    switch (flags) {
      case 0:
        k_gemm_bf16<0><<<g, 256, 0, stream>>>(A, Bt, bias, C, M, N, K, sA, sB, sC); break;
      case GF_OUT_BF16:
        k_gemm_bf16<GF_OUT_BF16><<<g, 256, 0, stream>>>(A, Bt, bias, C, M, N, K, sA, sB, sC); break;
      case GF_BIAS:
        k_gemm_bf16<GF_BIAS><<<g, 256, 0, stream>>>(A, Bt, bias, C, M, N, K, sA, sB, sC); break;
      case GF_BIAS | GF_OUT_BF16:
        k_gemm_bf16<GF_BIAS | GF_OUT_BF16><<<g, 256, 0, stream>>>(A, Bt, bias, C, M, N, K, sA, sB, sC); break;
      case GF_BIAS | GF_OUT_BF16 | GF_GELU:
        k_gemm_bf16<GF_BIAS | GF_OUT_BF16 | GF_GELU><<<g, 256, 0, stream>>>(A, Bt, bias, C, M, N, K, sA, sB, sC); break;
      default:
        k_gemm_bf16<0><<<g, 256, 0, stream>>>(A, Bt, bias, C, M, N, K, sA, sB, sC); break;
    }
  };

  // KVQ projections
  gemm(xb, WkT, bk, Kb, BL, DK, D, 0, 0, 0, 1, GF_BIAS | GF_OUT_BF16);
  gemm(xb, WvT, bv, Vf, BL, DV, D, 0, 0, 0, 1, GF_BIAS);
  gemm(xb, WqT, bq, Qb, BL, DK, D, 0, 0, 0, 1, GF_BIAS | GF_OUT_BF16);
  // memory read: (Q @ W) @ Wu + bu
  gemm(Qb, WmT, nullptr, QWb, L, DV, DK, (long)L * DK, (long)DK * DV, (long)L * DV, B,
       GF_OUT_BF16);
  gemm(QWb, WuT, bu, xmem, BL, D, DV, 0, 0, 0, 1, GF_BIAS);
  // attention block (pre-norm)
  k_layernorm_bf16<<<BL, 256, 0, stream>>>(xmem, ln1_g, ln1_b, hb, D);
  gemm(hb, WqkvT, nullptr, qkvb, BL, 3072, D, 0, 0, 0, 1, GF_OUT_BF16);
  k_flash_attn<<<dim3(L / 128, 16, B), 256, 0, stream>>>(qkvb, ob, L);
  gemm(ob, WoT, bo, xatt, BL, D, D, 0, 0, 0, 1, GF_BIAS);
  // MLP (pre-norm)
  k_layernorm_bf16<<<BL, 256, 0, stream>>>(xatt, ln2_g, ln2_b, h2b, D);
  gemm(h2b, W1T, b1, m1b, BL, MLP, D, 0, 0, 0, 1, GF_BIAS | GF_OUT_BF16 | GF_GELU);
  gemm(m1b, W2T, b2, (float*)d_out, BL, D, MLP, 0, 0, 0, 1, GF_BIAS);
  // Titans memory update
  gemm(Kb, WmT, nullptr, predf, L, DV, DK, (long)L * DK, (long)DK * DV, (long)L * DV, B, 0);
  k_subT_f32_bf16<<<(BL * DV + 255) / 256, 256, 0, stream>>>(predf, Vf, EbT, B, L, DV);
  k_transpose_bf16<<<(BL * DK + 255) / 256, 256, 0, stream>>>(Kb, KTb, B, L, DK);
  gemm(KTb, EbT, nullptr, gradf, DK, DV, L, (long)DK * L, (long)DV * L, (long)DK * DV, B, 0);
  float* out_W = (float*)d_out + (size_t)BL * D;
  float* out_S = out_W + (size_t)B * DK * DV;
  k_mem_update<<<(B * DK * DV + 255) / 256, 256, 0, stream>>>(
      gradf, Wmem, Smem, out_W, out_S, B * DK * DV);
}
